// SageNet_13202729468516
// MI455X (gfx1250) — matrix operand order repaired and checked by reading, not compiled, not run on hardware
//
#include <hip/hip_runtime.h>
#include <math.h>

#define N_NODES 100000
#define N_EDGES 600000
#define H_DIM   128
#define LH_DIM  256
#define C_DIM   8
#define THETA_F 1.57079632679489662f

typedef __attribute__((ext_vector_type(16))) _Float16 v16h;
typedef __attribute__((ext_vector_type(8)))  _Float16 v8h;
typedef __attribute__((ext_vector_type(4)))  _Float16 v4h;
typedef __attribute__((ext_vector_type(8)))  float    v8f;

union AFrag { v16h v; v8h h[2]; };

// ---------- float <-> orderable-uint helpers for atomic float max ----------
__device__ __forceinline__ unsigned flipf(float f) {
    unsigned u = __float_as_uint(f);
    return (u & 0x80000000u) ? ~u : (u | 0x80000000u);
}
__device__ __forceinline__ float unflipf(unsigned u) {
    u = (u & 0x80000000u) ? (u & 0x7FFFFFFFu) : ~u;
    return __uint_as_float(u);
}

// ---------- pass 1: fused normalize statistics ----------
// su[0]=max x, su[1]=max y, su[2]=max(-x), su[3]=max(-y),
// su[4]=max(c*x-s*y), su[5]=max(s*x+c*y), su[6]=max area ; ss[0]=sum x, ss[1]=sum y
__global__ void stats_kernel(const float* __restrict__ x,
                             unsigned* __restrict__ su, float* __restrict__ ss) {
    int tid = blockIdx.x * blockDim.x + threadIdx.x;
    int stride = gridDim.x * blockDim.x;
    const float c = cosf(THETA_F), s = sinf(THETA_F);
    float m0 = -3.4e38f, m1 = -3.4e38f, m2 = -3.4e38f, m3 = -3.4e38f;
    float m4 = -3.4e38f, m5 = -3.4e38f, m6 = -3.4e38f;
    float s0 = 0.f, s1 = 0.f;
    for (int i = tid; i < N_NODES; i += stride) {
        float px = x[i * 3 + 0], py = x[i * 3 + 1], pa = x[i * 3 + 2];
        m0 = fmaxf(m0, px);  m1 = fmaxf(m1, py);
        m2 = fmaxf(m2, -px); m3 = fmaxf(m3, -py);
        m4 = fmaxf(m4, c * px - s * py);
        m5 = fmaxf(m5, s * px + c * py);
        m6 = fmaxf(m6, pa);
        s0 += px; s1 += py;
    }
    atomicMax(&su[0], flipf(m0)); atomicMax(&su[1], flipf(m1));
    atomicMax(&su[2], flipf(m2)); atomicMax(&su[3], flipf(m3));
    atomicMax(&su[4], flipf(m4)); atomicMax(&su[5], flipf(m5));
    atomicMax(&su[6], flipf(m6));
    atomicAdd(&ss[0], s0); atomicAdd(&ss[1], s1);
}

// ---------- pass 2: apply normalization ----------
__global__ void normalize_kernel(const float* __restrict__ x,
                                 const unsigned* __restrict__ su,
                                 const float* __restrict__ ss,
                                 float* __restrict__ h0) {
    int i = blockIdx.x * blockDim.x + threadIdx.x;
    if (i >= N_NODES) return;
    float mx0 = unflipf(su[0]), mx1 = unflipf(su[1]);
    float mn0 = -unflipf(su[2]), mn1 = -unflipf(su[3]);
    float mxr0 = unflipf(su[4]), mxr1 = unflipf(su[5]);
    float mxa = unflipf(su[6]);
    bool rot = (mx1 - mn1) > (mx0 - mn0);
    const float c = cosf(THETA_F), s = sinf(THETA_F);
    float sum0 = ss[0], sum1 = ss[1];
    float invN = 1.0f / (float)N_NODES;
    float mean0, mean1, den0, den1;
    if (rot) { mean0 = (c * sum0 - s * sum1) * invN; mean1 = (s * sum0 + c * sum1) * invN;
               den0 = mxr0; den1 = mxr1; }
    else     { mean0 = sum0 * invN; mean1 = sum1 * invN; den0 = mx0; den1 = mx1; }
    float px = x[i * 3 + 0], py = x[i * 3 + 1], pa = x[i * 3 + 2];
    float cx = rot ? (c * px - s * py) : px;
    float cy = rot ? (s * px + c * py) : py;
    h0[i * 3 + 0] = (cx - mean0) / den0;
    h0[i * 3 + 1] = (cy - mean1) / den1;
    h0[i * 3 + 2] = pa / mxa;
}

// ---------- layer-1 scatter: deg + 3-wide message sum ----------
__global__ void edge1_kernel(const int* __restrict__ ei, const float* __restrict__ h0,
                             float* __restrict__ deg, float* __restrict__ agg1) {
    int e = blockIdx.x * blockDim.x + threadIdx.x;
    if (e >= N_EDGES) return;
    int src = ei[e], dst = ei[N_EDGES + e];
    atomicAdd(&deg[dst], 1.0f);
    atomicAdd(&agg1[dst * 3 + 0], h0[src * 3 + 0]);
    atomicAdd(&agg1[dst * 3 + 1], h0[src * 3 + 1]);
    atomicAdd(&agg1[dst * 3 + 2], h0[src * 3 + 2]);
}

// ---------- layer-1 dense part (K=3, scalar; writes f16 activations) ----------
__global__ void layer1_kernel(const float* __restrict__ h0, const float* __restrict__ agg1,
                              const float* __restrict__ deg,
                              const float* __restrict__ Wl1, const float* __restrict__ bl1,
                              const float* __restrict__ Wr1, _Float16* __restrict__ h1) {
    long long idx = (long long)blockIdx.x * blockDim.x + threadIdx.x;
    if (idx >= (long long)N_NODES * H_DIM) return;
    int i = (int)(idx >> 7);
    int ch = (int)(idx & 127);
    float d = fmaxf(deg[i], 1.0f);
    float invd = 1.0f / d;
    float a0 = agg1[i * 3 + 0] * invd, a1 = agg1[i * 3 + 1] * invd, a2 = agg1[i * 3 + 2] * invd;
    float x0 = h0[i * 3 + 0], x1 = h0[i * 3 + 1], x2 = h0[i * 3 + 2];
    float v = bl1[ch]
            + a0 * Wl1[ch * 3 + 0] + a1 * Wl1[ch * 3 + 1] + a2 * Wl1[ch * 3 + 2]
            + x0 * Wr1[ch * 3 + 0] + x1 * Wr1[ch * 3 + 1] + x2 * Wr1[ch * 3 + 2];
    h1[(size_t)i * H_DIM + ch] = (_Float16)tanhf(v);
}

// ---------- layer-2 scatter: warp per edge, 128 channels, f32 atomics ----------
__global__ void edge2_kernel(const int* __restrict__ ei, const _Float16* __restrict__ h1,
                             float* __restrict__ agg2) {
    long long gt = (long long)blockIdx.x * blockDim.x + threadIdx.x;
    int e = (int)(gt >> 5);
    int lane = (int)(gt & 31);
    if (e >= N_EDGES) return;
    int src = ei[e], dst = ei[N_EDGES + e];
    const _Float16* hrow = h1 + (size_t)src * H_DIM + lane * 4;
    float* arow = agg2 + (size_t)dst * H_DIM + lane * 4;
    v4h hv = *(const v4h*)hrow;
    atomicAdd(arow + 0, (float)hv[0]);
    atomicAdd(arow + 1, (float)hv[1]);
    atomicAdd(arow + 2, (float)hv[2]);
    atomicAdd(arow + 3, (float)hv[3]);
}

// ---------- agg2: divide by degree, convert to f16 ----------
__global__ void agg2cvt_kernel(const float* __restrict__ agg2, const float* __restrict__ deg,
                               _Float16* __restrict__ agg2h) {
    long long idx = (long long)blockIdx.x * blockDim.x + threadIdx.x;
    if (idx >= (long long)N_NODES * H_DIM) return;
    int i = (int)(idx >> 7);
    float d = fmaxf(deg[i], 1.0f);
    agg2h[idx] = (_Float16)(agg2[idx] / d);
}

// ---------- generic f32 -> f16 convert (for weights) ----------
__global__ void cvt_kernel(const float* __restrict__ in, _Float16* __restrict__ out, int n) {
    int i = blockIdx.x * blockDim.x + threadIdx.x;
    if (i < n) out[i] = (_Float16)in[i];
}

// ---------- WMMA GEMM: Out = tanh(X @ W^T [+ X2 @ W2^T] + bias), f16 in/out, f32 accum ----------
// Wave handles a 16(M) x 64(N) tile. A/B fragment lane layouts per CDNA5 ISA 7.12.2:
//   A (16x32 f16): lane l holds row m=l&15; halves 0..7 -> K = kc + 8*(l>>4) + h,
//                  halves 8..15 -> K = kc + 16 + 8*(l>>4) + (h&7)  (two contiguous 8-half runs)
//   B (32x16 f16): lane l holds col n=l&15 (== output o); B[k,n]=W[o,k] so lane reads
//                  weight row o with the same K mapping as A (two 8-half runs at kc + 8*(l>>4) and +16)
// KDIM/NPASS are compile-time so the kc loop fully unrolls: per chunk we batch A + 4 B
// fragment loads (one s_clause group), then issue 4 back-to-back v_wmma, letting the
// next chunk's loads overlap the current chunk's WMMAs.
template <int KDIM, int NPASS>
__global__ void __launch_bounds__(128) gemm_wmma(
    const _Float16* __restrict__ X,  const _Float16* __restrict__ W,
    const _Float16* __restrict__ X2, const _Float16* __restrict__ W2,
    const float* __restrict__ bias, _Float16* __restrict__ Out,
    int nrows, int O, int colGroups) {
    const int wave = threadIdx.x >> 5;
    const int lane = threadIdx.x & 31;
    const int gw = blockIdx.x * (blockDim.x >> 5) + wave;
    const int rowTile = gw / colGroups;
    const int colGroup = gw - rowTile * colGroups;
    const int row0 = rowTile * 16;
    if (row0 >= nrows) return;                 // wave-uniform: EXEC stays all-ones
    const int col0 = colGroup * 64;
    const int laneHi = lane >> 4;              // 0 or 1
    const int lane15 = lane & 15;

    v8f acc[4];
#pragma unroll
    for (int t = 0; t < 4; ++t) acc[t] = v8f{0.f, 0.f, 0.f, 0.f, 0.f, 0.f, 0.f, 0.f};

    const size_t arowOff = (size_t)(row0 + lane15) * KDIM + (laneHi << 3);
    const size_t bkOff = (size_t)(laneHi << 3);

#pragma unroll
    for (int pass = 0; pass < NPASS; ++pass) {
        const _Float16* __restrict__ Xp = pass ? X2 : X;
        const _Float16* __restrict__ Wp = pass ? W2 : W;
#pragma unroll
        for (int kc = 0; kc < KDIM; kc += 32) {
            AFrag a, b[4];
            const _Float16* ap = Xp + arowOff + kc;
            a.h[0] = *(const v8h*)(ap);
            a.h[1] = *(const v8h*)(ap + 16);
#pragma unroll
            for (int t = 0; t < 4; ++t) {
                const _Float16* bp = Wp + (size_t)(col0 + t * 16 + lane15) * KDIM + kc + bkOff;
                b[t].h[0] = *(const v8h*)(bp);
                b[t].h[1] = *(const v8h*)(bp + 16);
            }
#pragma unroll
            for (int t = 0; t < 4; ++t) {
                acc[t] = __builtin_amdgcn_wmma_f32_16x16x32_f16(
                    false, a.v, false, b[t].v, (short)0, acc[t], false, false);
            }
        }
    }

    // C/D layout: lane l, VGPR r -> M = row0 + r + 8*(l>>4), N(=o) = col0 + t*16 + (l&15)
#pragma unroll
    for (int t = 0; t < 4; ++t) {
        int o = col0 + t * 16 + lane15;
        float bv = bias[o];
#pragma unroll
        for (int r = 0; r < 8; ++r) {
            int m = row0 + r + (laneHi << 3);
            float v = acc[t][r] + bv;
            Out[(size_t)m * O + o] = (_Float16)tanhf(v);
        }
    }
}

// ---------- head: warp per node, 8x 256-dot + softmax ----------
__global__ void __launch_bounds__(256) head_kernel(const _Float16* __restrict__ H3,
                                                   const float* __restrict__ W2,
                                                   const float* __restrict__ b2,
                                                   float* __restrict__ out) {
    long long gt = (long long)blockIdx.x * blockDim.x + threadIdx.x;
    int node = (int)(gt >> 5);
    int lane = (int)(gt & 31);
    if (node >= N_NODES) return;
    const _Float16* hrow = H3 + (size_t)node * LH_DIM;
    float hv[8];
#pragma unroll
    for (int j = 0; j < 8; ++j) hv[j] = (float)hrow[lane + j * 32];
    float z[C_DIM];
#pragma unroll
    for (int c = 0; c < C_DIM; ++c) {
        float p = 0.f;
#pragma unroll
        for (int j = 0; j < 8; ++j) p += hv[j] * W2[c * LH_DIM + lane + j * 32];
        for (int off = 16; off > 0; off >>= 1) p += __shfl_down(p, off, 32);
        z[c] = p;
    }
    if (lane == 0) {
        float zz[C_DIM];
        float mx = -3.4e38f;
#pragma unroll
        for (int c = 0; c < C_DIM; ++c) { zz[c] = z[c] + b2[c]; mx = fmaxf(mx, zz[c]); }
        float ssum = 0.f;
#pragma unroll
        for (int c = 0; c < C_DIM; ++c) { zz[c] = expf(zz[c] - mx); ssum += zz[c]; }
        float inv = 1.0f / ssum;
#pragma unroll
        for (int c = 0; c < C_DIM; ++c) out[(size_t)node * C_DIM + c] = zz[c] * inv;
    }
}

extern "C" void kernel_launch(void* const* d_in, const int* in_sizes, int n_in,
                              void* d_out, int out_size, void* d_ws, size_t ws_size,
                              hipStream_t stream) {
    const float* x   = (const float*)d_in[0];
    const int*   ei  = (const int*)d_in[1];
    const float* Wl1 = (const float*)d_in[2];
    const float* bl1 = (const float*)d_in[3];
    const float* Wr1 = (const float*)d_in[4];
    const float* Wl2 = (const float*)d_in[5];
    const float* bl2 = (const float*)d_in[6];
    const float* Wr2 = (const float*)d_in[7];
    const float* W1  = (const float*)d_in[8];
    const float* b1  = (const float*)d_in[9];
    const float* W2  = (const float*)d_in[10];
    const float* b2  = (const float*)d_in[11];
    float* out = (float*)d_out;

    char* ws = (char*)d_ws;
    size_t off = 0;
    auto alloc = [&](size_t bytes) -> char* {
        char* p = ws + off;
        off += (bytes + 255) & ~(size_t)255;
        return p;
    };

    unsigned*  stats  = (unsigned*)alloc(256);               // su[0..6] flipped-max, ss[0..1] sums
    float*     statsS = (float*)(stats + 8);
    float*     h0     = (float*)alloc((size_t)N_NODES * 3 * 4);
    float*     deg    = (float*)alloc((size_t)N_NODES * 4);
    float*     agg1   = (float*)alloc((size_t)N_NODES * 3 * 4);
    _Float16*  h1h    = (_Float16*)alloc((size_t)N_NODES * H_DIM * 2);
    float*     agg2   = (float*)alloc((size_t)N_NODES * H_DIM * 4);   // reused as h3h (same bytes)
    _Float16*  h3h    = (_Float16*)agg2;                              // N x 256 f16 == N x 128 f32
    _Float16*  agg2h  = (_Float16*)alloc((size_t)N_NODES * H_DIM * 2);
    _Float16*  h2h    = (_Float16*)alloc((size_t)N_NODES * H_DIM * 2);
    _Float16*  wl2h   = (_Float16*)alloc((size_t)H_DIM * H_DIM * 2);
    _Float16*  wr2h   = (_Float16*)alloc((size_t)H_DIM * H_DIM * 2);
    _Float16*  w1h    = (_Float16*)alloc((size_t)LH_DIM * H_DIM * 2);

    // zero-init accumulators (flipped-max init 0 == below flip(-inf); sums 0.0f)
    hipMemsetAsync(stats, 0, 256, stream);
    hipMemsetAsync(deg, 0, (size_t)N_NODES * 4, stream);
    hipMemsetAsync(agg1, 0, (size_t)N_NODES * 3 * 4, stream);
    hipMemsetAsync(agg2, 0, (size_t)N_NODES * H_DIM * 4, stream);

    // normalize
    stats_kernel<<<8, 256, 0, stream>>>(x, stats, statsS);
    normalize_kernel<<<(N_NODES + 255) / 256, 256, 0, stream>>>(x, stats, statsS, h0);

    // SAGE layer 1
    edge1_kernel<<<(N_EDGES + 255) / 256, 256, 0, stream>>>(ei, h0, deg, agg1);
    layer1_kernel<<<((size_t)N_NODES * H_DIM + 255) / 256, 256, 0, stream>>>(
        h0, agg1, deg, Wl1, bl1, Wr1, h1h);

    // SAGE layer 2 scatter + normalize
    edge2_kernel<<<((size_t)N_EDGES * 32 + 255) / 256, 256, 0, stream>>>(ei, h1h, agg2);
    agg2cvt_kernel<<<((size_t)N_NODES * H_DIM + 255) / 256, 256, 0, stream>>>(agg2, deg, agg2h);

    // weight conversions to f16
    cvt_kernel<<<(H_DIM * H_DIM + 255) / 256, 256, 0, stream>>>(Wl2, wl2h, H_DIM * H_DIM);
    cvt_kernel<<<(H_DIM * H_DIM + 255) / 256, 256, 0, stream>>>(Wr2, wr2h, H_DIM * H_DIM);
    cvt_kernel<<<(LH_DIM * H_DIM + 255) / 256, 256, 0, stream>>>(W1, w1h, LH_DIM * H_DIM);

    // layer 2 dense: h2 = tanh(agg2n @ Wl2^T + h1 @ Wr2^T + bl2)   (fused dual-GEMM, WMMA)
    {
        const int colGroups = H_DIM / 64;                 // 2
        const int waves = (N_NODES / 16) * colGroups;     // 12500
        gemm_wmma<H_DIM, 2><<<waves / 4, 128, 0, stream>>>(agg2h, wl2h, h1h, wr2h, bl2, h2h,
                                                           N_NODES, H_DIM, colGroups);
    }
    // layer 3 dense: h3 = tanh(h2 @ W1^T + b1)  (WMMA)
    {
        const int colGroups = LH_DIM / 64;                // 4
        const int waves = (N_NODES / 16) * colGroups;     // 25000
        gemm_wmma<H_DIM, 1><<<waves / 4, 128, 0, stream>>>(h2h, w1h, nullptr, nullptr, b1, h3h,
                                                           N_NODES, LH_DIM, colGroups);
    }
    // head + softmax
    head_kernel<<<((size_t)N_NODES * 32 + 255) / 256, 256, 0, stream>>>(h3h, W2, b2, out);
}
